// OuterProductMean_9268539425322
// MI455X (gfx1250) — matrix operand order repaired and checked by reading, not compiled, not run on hardware
//
#include <hip/hip_runtime.h>
#include <hip/hip_bf16.h>

#define IN_C  256
#define MID_C 32
#define OUT_C 128
#define MDIM  128
#define NDIM  256
#define POS   (MDIM * NDIM)   // 32768 (m,n) positions

typedef __bf16 bf16_t;
typedef bf16_t v16bf __attribute__((ext_vector_type(16)));
typedef float  v8f   __attribute__((ext_vector_type(8)));

union FragB {
    uint4 q[2];
    v16bf v;
};

static __device__ __forceinline__ unsigned short f2bf(float f) {
    bf16_t h = (bf16_t)f;
    return __builtin_bit_cast(unsigned short, h);
}

static __device__ __forceinline__ v8f wmma_bf16(v16bf a, v16bf b, v8f c) {
    return __builtin_amdgcn_wmma_f32_16x16x32_bf16(false, a, false, b, (short)0, c,
                                                   false, false);
}

// -------------------------------------------------------------------------
// Kernel 1: LayerNorm over channels, output xn in bf16 [POS][IN_C].
// One wave per row; lane handles 8 contiguous channels; shuffle reduction.
// -------------------------------------------------------------------------
__global__ void __launch_bounds__(256) ln_kernel(const float* __restrict__ x,
                                                 const float* __restrict__ gamma,
                                                 const float* __restrict__ beta,
                                                 unsigned short* __restrict__ xn) {
    const int lane = threadIdx.x & 31;
    const int wv   = threadIdx.x >> 5;
    const int row  = blockIdx.x * 8 + wv;           // 4096 blocks * 8 waves = 32768 rows
    const int c0   = lane * 8;

    const float4* xr = reinterpret_cast<const float4*>(x + (size_t)row * IN_C + c0);
    float4 v0 = xr[0];
    float4 v1 = xr[1];

    float s = v0.x + v0.y + v0.z + v0.w + v1.x + v1.y + v1.z + v1.w;
    float q = v0.x * v0.x + v0.y * v0.y + v0.z * v0.z + v0.w * v0.w +
              v1.x * v1.x + v1.y * v1.y + v1.z * v1.z + v1.w * v1.w;
    for (int m = 16; m > 0; m >>= 1) {
        s += __shfl_xor(s, m, 32);
        q += __shfl_xor(q, m, 32);
    }
    const float mu   = s * (1.0f / IN_C);
    const float var  = q * (1.0f / IN_C) - mu * mu;
    const float rstd = rsqrtf(var + 1e-5f);

    const float4 g0 = reinterpret_cast<const float4*>(gamma + c0)[0];
    const float4 g1 = reinterpret_cast<const float4*>(gamma + c0)[1];
    const float4 b0 = reinterpret_cast<const float4*>(beta + c0)[0];
    const float4 b1 = reinterpret_cast<const float4*>(beta + c0)[1];

    float o[8];
    o[0] = (v0.x - mu) * rstd * g0.x + b0.x;
    o[1] = (v0.y - mu) * rstd * g0.y + b0.y;
    o[2] = (v0.z - mu) * rstd * g0.z + b0.z;
    o[3] = (v0.w - mu) * rstd * g0.w + b0.w;
    o[4] = (v1.x - mu) * rstd * g1.x + b1.x;
    o[5] = (v1.y - mu) * rstd * g1.y + b1.y;
    o[6] = (v1.z - mu) * rstd * g1.z + b1.z;
    o[7] = (v1.w - mu) * rstd * g1.w + b1.w;

    uint4 pk;
    pk.x = (unsigned)f2bf(o[0]) | ((unsigned)f2bf(o[1]) << 16);
    pk.y = (unsigned)f2bf(o[2]) | ((unsigned)f2bf(o[3]) << 16);
    pk.z = (unsigned)f2bf(o[4]) | ((unsigned)f2bf(o[5]) << 16);
    pk.w = (unsigned)f2bf(o[6]) | ((unsigned)f2bf(o[7]) << 16);
    *reinterpret_cast<uint4*>(xn + (size_t)row * IN_C + c0) = pk;
}

// -------------------------------------------------------------------------
// Kernel 2: swizzle weights into WMMA B-fragment layout (bf16).
// B-frag layout: lane holds column n = lane&15, halves h=0..15 hold
// K = kbase + (lane>>4)*16 + h (contiguous).
// wcf: [kstep(8)][coltile(4)][lane(32)][half(16)]   (Wl|Wr, 64 cols)
// wff: [kstep(32)][otile(8)][lane(32)][half(16)]    (Wf, 1024x128)
// -------------------------------------------------------------------------
__global__ void prep_kernel(const float* __restrict__ Wl, const float* __restrict__ Wr,
                            const float* __restrict__ Wf,
                            unsigned short* __restrict__ wcf,
                            unsigned short* __restrict__ wff) {
    const int idx = blockIdx.x * blockDim.x + threadIdx.x;
    if (idx < 16384) {
        const int h    = idx & 15;
        const int lane = (idx >> 4) & 31;
        const int ct   = (idx >> 9) & 3;
        const int s    = idx >> 11;
        const int k    = s * 32 + (lane >> 4) * 16 + h;
        const int col  = ct * 16 + (lane & 15);
        const float v  = (col < MID_C) ? Wl[k * MID_C + col] : Wr[k * MID_C + col - MID_C];
        wcf[idx] = f2bf(v);
    } else if (idx < 16384 + 131072) {
        const int j    = idx - 16384;
        const int h    = j & 15;
        const int lane = (j >> 4) & 31;
        const int ot   = (j >> 9) & 7;
        const int s    = j >> 12;
        const int k    = s * 32 + (lane >> 4) * 16 + h;
        const int o    = ot * 16 + (lane & 15);
        wff[j] = f2bf(Wf[k * OUT_C + o]);
    }
}

// -------------------------------------------------------------------------
// Kernel 3: projections. LR[32768 x 64] = xn[32768 x 256] @ [Wl|Wr], + bias.
// Outputs transposed over m: leftT[n][x][m], rightT[n][y][m] (bf16),
// so the fused kernel gets m-contiguous 128-bit fragment loads.
// -------------------------------------------------------------------------
__global__ void __launch_bounds__(256) proj_kernel(const unsigned short* __restrict__ xn,
                                                   const unsigned short* __restrict__ wcf,
                                                   const float* __restrict__ bl,
                                                   const float* __restrict__ br,
                                                   unsigned short* __restrict__ leftT,
                                                   unsigned short* __restrict__ rightT) {
    const int lane    = threadIdx.x & 31;
    const int wv      = threadIdx.x >> 5;
    const int t       = blockIdx.x * 8 + wv;        // 1024 blocks * 8 = 8192 tiles
    const int rowTile = t >> 2;                     // 0..2047
    const int colTile = t & 3;                      // 0..3 (16 cols each)
    const int laneLo  = lane & 15;
    const int laneHi  = lane >> 4;

    const unsigned short* arow = xn + (size_t)(rowTile * 16 + laneLo) * IN_C;
    v8f acc = {};
    for (int s = 0; s < 8; ++s) {
        FragB a, b;
        a.q[0] = *reinterpret_cast<const uint4*>(arow + s * 32 + laneHi * 16);
        a.q[1] = *reinterpret_cast<const uint4*>(arow + s * 32 + laneHi * 16 + 8);
        const uint4* bp =
            reinterpret_cast<const uint4*>(wcf + ((s * 4 + colTile) * 32 + lane) * 16);
        b.q[0] = bp[0];
        b.q[1] = bp[1];
        acc = wmma_bf16(a.v, b.v, acc);
    }

    const int   col  = colTile * 16 + laneLo;                // 0..63
    const float bias = (colTile < 2) ? bl[col] : br[col - MID_C];
    unsigned short* dst  = (colTile < 2) ? leftT : rightT;
    const int       feat = (colTile < 2) ? col : (col - MID_C);

    for (int r = 0; r < 8; ++r) {
        const int pos = rowTile * 16 + r + laneHi * 8;
        const int mm  = pos >> 8;       // m index (n is fastest: pos = m*256 + n)
        const int nn  = pos & 255;      // n index
        dst[((size_t)nn * MID_C + feat) * MDIM + mm] = f2bf(acc[r] + bias);
    }
}

// -------------------------------------------------------------------------
// Kernel 4: fused outer-product (sum over m) + final 1024->128 linear.
// Block = (i, 16 j's). Phase A: op[j,x,y] = sum_m L[m,i,x] R[m,j,y] via WMMA
// (K=m=128), bf16 result staged in LDS. Phase B: out[j,o] =
// (sum_{x,y} op[j,x,y] * Wf[x,y,o] + bf[o]) / 128.001 via WMMA (K=1024).
// -------------------------------------------------------------------------
__global__ void __launch_bounds__(256) fused_kernel(const unsigned short* __restrict__ leftT,
                                                    const unsigned short* __restrict__ rightT,
                                                    const unsigned short* __restrict__ wff,
                                                    const float* __restrict__ bfv,
                                                    float* __restrict__ out) {
    __shared__ unsigned short opLds[16 * 32 * 32];  // [j][x][y] bf16, 32KB

    const int lane   = threadIdx.x & 31;
    const int wv     = threadIdx.x >> 5;            // 8 waves
    const int i      = blockIdx.x >> 4;             // 0..255
    const int jt     = blockIdx.x & 15;             // j tile of 16
    const int laneLo = lane & 15;
    const int laneHi = lane >> 4;

    // ---- Phase A: GEMM1, 32(x) x 512(j,y) tile, K = m = 128 ----
    FragB afr[2][4];                                // shared A (left) fragments
    for (int xh = 0; xh < 2; ++xh)
        for (int s = 0; s < 4; ++s) {
            const unsigned short* ap =
                leftT + ((size_t)(i * MID_C + xh * 16 + laneLo)) * MDIM + s * 32 + laneHi * 16;
            afr[xh][s].q[0] = *reinterpret_cast<const uint4*>(ap);
            afr[xh][s].q[1] = *reinterpret_cast<const uint4*>(ap + 8);
        }

    for (int l = 2 * wv; l < 2 * wv + 2; ++l) {     // 2 local j per wave
        const int jg = jt * 16 + l;
        for (int yh = 0; yh < 2; ++yh) {
            const int y = yh * 16 + laneLo;
            const unsigned short* brow = rightT + ((size_t)(jg * MID_C + y)) * MDIM;
            v8f acc0 = {};
            v8f acc1 = {};
            for (int s = 0; s < 4; ++s) {
                FragB b;
                const uint4* bp = reinterpret_cast<const uint4*>(brow + s * 32 + laneHi * 16);
                b.q[0] = bp[0];
                b.q[1] = bp[1];
                acc0 = wmma_bf16(afr[0][s].v, b.v, acc0);
                acc1 = wmma_bf16(afr[1][s].v, b.v, acc1);
            }
            for (int r = 0; r < 8; ++r) {
                const int x0 = r + laneHi * 8;
                opLds[(l * 32 + x0) * 32 + y]        = f2bf(acc0[r]);
                opLds[(l * 32 + 16 + x0) * 32 + y]   = f2bf(acc1[r]);
            }
        }
    }
    __syncthreads();

    // ---- Phase B: GEMM2, 16(j) x 16(o per wave), K = (x,y) = 1024 ----
    v8f acc = {};
    for (int s = 0; s < 32; ++s) {
        FragB a, b;
        // A rows = j (lane&15); K chunk: x = s, y = laneHi*16 + e (e = 0..15)
        const unsigned short* ap = &opLds[(laneLo * 32 + s) * 32 + laneHi * 16];
        a.q[0] = *reinterpret_cast<const uint4*>(ap);
        a.q[1] = *reinterpret_cast<const uint4*>(ap + 8);
        const uint4* bp = reinterpret_cast<const uint4*>(wff + ((s * 8 + wv) * 32 + lane) * 16);
        if (s + 1 < 32)
            __builtin_prefetch(wff + (((s + 1) * 8 + wv) * 32 + lane) * 16, 0, 1);
        b.q[0] = bp[0];
        b.q[1] = bp[1];
        acc = wmma_bf16(a.v, b.v, acc);
    }

    const float inv = 1.0f / 128.001f;              // / (M + 0.001)
    const int   o   = wv * 16 + laneLo;
    const float bo  = bfv[o];
    for (int r = 0; r < 8; ++r) {
        const int jg = jt * 16 + r + laneHi * 8;
        out[((size_t)(i * NDIM + jg)) * OUT_C + o] = (acc[r] + bo) * inv;
    }
}

// -------------------------------------------------------------------------
// Host launcher
// -------------------------------------------------------------------------
extern "C" void kernel_launch(void* const* d_in, const int* in_sizes, int n_in,
                              void* d_out, int out_size, void* d_ws, size_t ws_size,
                              hipStream_t stream) {
    const float* x1d   = (const float*)d_in[0];
    const float* gamma = (const float*)d_in[1];
    const float* beta  = (const float*)d_in[2];
    const float* Wl    = (const float*)d_in[3];
    const float* bl    = (const float*)d_in[4];
    const float* Wr    = (const float*)d_in[5];
    const float* br    = (const float*)d_in[6];
    const float* Wf    = (const float*)d_in[7];
    const float* bf    = (const float*)d_in[8];
    float*       out   = (float*)d_out;

    // Workspace layout (bytes)
    char* ws = (char*)d_ws;
    unsigned short* xn     = (unsigned short*)(ws);                 // 32768*256*2 = 16 MB
    unsigned short* leftT  = (unsigned short*)(ws + 16777216);      // 256*32*128*2 = 2 MB
    unsigned short* rightT = (unsigned short*)(ws + 16777216 + 2097152);
    unsigned short* wcf    = (unsigned short*)(ws + 16777216 + 2 * 2097152);          // 32 KB
    unsigned short* wff    = (unsigned short*)(ws + 16777216 + 2 * 2097152 + 32768);  // 256 KB

    ln_kernel<<<POS / 8, 256, 0, stream>>>(x1d, gamma, beta, xn);
    prep_kernel<<<(16384 + 131072) / 256, 256, 0, stream>>>(Wl, Wr, Wf, wcf, wff);
    proj_kernel<<<1024, 256, 0, stream>>>(xn, wcf, bl, br, leftT, rightT);
    fused_kernel<<<NDIM * 16, 256, 0, stream>>>(leftT, rightT, wff, bf, out);
}
